// GCN_GRU_69028714381734
// MI455X (gfx1250) — matrix operand order repaired and checked by reading, not compiled, not run on hardware
//
#include <hip/hip_runtime.h>

// ---------------------------------------------------------------------------
// AGCRN-style GCN-GRU for MI455X (gfx1250, wave32, WMMA bf16 16x16x32).
//
// B=32 T=12 N=2000 D_IN=2 H=64 K=2 E=10 L=2.
// Dominant cost: per-timestep aggregation A[2000x2000] @ X[2000x(B*C)]
// (~1.2 TFLOP total) -> compute bound on the WMMA units; A (8MB bf16) is
// L2-resident (192MB).  bf16 inputs / f32 accumulate via
// v_wmma_f32_16x16x32_bf16.  Tile movement uses the CDNA5 async path:
// global_load_async_to_lds_b128 + s_wait_asynccnt, double-buffered so global
// traffic overlaps the WMMA stream.
// ---------------------------------------------------------------------------

#define NN   2000          // nodes
#define NPAD 2016          // node dim padded to multiple of 32 (WMMA K)
#define MPAD 2048          // adjacency rows padded to multiple of 128 (tile M)
#define BB   32            // batch
#define TT   12            // timesteps
#define DIN  2
#define HH   64
#define EE   10

typedef __bf16 bf16;
typedef __attribute__((ext_vector_type(16))) __bf16 v16bf;
typedef __attribute__((ext_vector_type(8)))  float  v8f;

union Frag { v16bf v; unsigned int u[8]; };

__device__ __forceinline__ float sigmoidf_(float x) { return 1.0f / (1.0f + __expf(-x)); }

// NOTE: generic pointers to __shared__ have the LDS offset in their low 32
// bits on gfx1250 (LDS aperture truncation, ISA 10.2) - used for async dests.
__device__ __forceinline__ unsigned lds_off(const void* p) {
    return (unsigned)(size_t)p;
}
__device__ __forceinline__ void async_b128(unsigned lds, const void* gp) {
    asm volatile("global_load_async_to_lds_b128 %0, %1, off"
                 :: "v"(lds), "v"((unsigned long long)(size_t)gp) : "memory");
}

// ---------------------------------------------------------------------------
// Kernel 1: A = softmax(relu(emb @ emb^T)) row-wise, bf16, zero-padded to
// [MPAD][NPAD] (WMMA-friendly shape).
// ---------------------------------------------------------------------------
__global__ __launch_bounds__(256) void adj_softmax_kernel(
    const float* __restrict__ emb, bf16* __restrict__ Abf)
{
    const int n   = blockIdx.x;       // 0..MPAD-1
    const int tid = threadIdx.x;      // 256 threads
    __shared__ float row[NN];
    __shared__ float red[256];

    if (n >= NN) {                    // padding rows: all zero
        for (int m = tid; m < NPAD; m += 256) Abf[(size_t)n * NPAD + m] = (bf16)0.0f;
        return;
    }
    float en[EE];
#pragma unroll
    for (int e = 0; e < EE; ++e) en[e] = emb[n * EE + e];

    float lmax = 0.0f;                // relu output >= 0
    for (int m = tid; m < NN; m += 256) {
        float s = 0.0f;
#pragma unroll
        for (int e = 0; e < EE; ++e) s += en[e] * emb[m * EE + e];
        s = s > 0.0f ? s : 0.0f;
        row[m] = s;
        lmax = fmaxf(lmax, s);
    }
    red[tid] = lmax; __syncthreads();
    for (int off = 128; off > 0; off >>= 1) {
        if (tid < off) red[tid] = fmaxf(red[tid], red[tid + off]);
        __syncthreads();
    }
    const float vmax = red[0]; __syncthreads();

    float lsum = 0.0f;
    for (int m = tid; m < NN; m += 256) {
        float ev = __expf(row[m] - vmax);
        row[m] = ev; lsum += ev;
    }
    red[tid] = lsum; __syncthreads();
    for (int off = 128; off > 0; off >>= 1) {
        if (tid < off) red[tid] += red[tid + off];
        __syncthreads();
    }
    const float inv = 1.0f / red[0];
    for (int m = tid; m < NPAD; m += 256)
        Abf[(size_t)n * NPAD + m] = (bf16)((m < NN) ? row[m] * inv : 0.0f);
}

// ---------------------------------------------------------------------------
// Kernel 2: per-node weights  W[n][k*Cp+c][o] = sum_e emb[n,e]*pool[e,k,c,o]
// (zero for padded channels c>=din).  pool: [E][2][din][O], out bf16.
// ---------------------------------------------------------------------------
__global__ void make_w_kernel(const float* __restrict__ emb, const float* __restrict__ pool,
                              bf16* __restrict__ Wout, int din, int Cp, int O)
{
    const size_t total  = (size_t)NN * 2 * Cp * O;
    const size_t stride = (size_t)gridDim.x * blockDim.x;
    for (size_t idx = (size_t)blockIdx.x * blockDim.x + threadIdx.x; idx < total; idx += stride) {
        int o = (int)(idx % O);
        size_t r2 = idx / O;
        int q = (int)(r2 % (2 * Cp));
        int n = (int)(r2 / (2 * Cp));
        int k = q / Cp, c = q % Cp;
        float a = 0.0f;
        if (c < din) {
            size_t pb = ((size_t)k * din + c) * O + o;
#pragma unroll
            for (int e = 0; e < EE; ++e)
                a += emb[n * EE + e] * pool[(size_t)e * (2 * (size_t)din * O) + pb];
        }
        Wout[idx] = (bf16)a;
    }
}

__global__ void make_bias_kernel(const float* __restrict__ emb, const float* __restrict__ bp,
                                 float* __restrict__ bias, int O)
{
    const size_t total  = (size_t)NN * O;
    const size_t stride = (size_t)gridDim.x * blockDim.x;
    for (size_t idx = (size_t)blockIdx.x * blockDim.x + threadIdx.x; idx < total; idx += stride) {
        int o = (int)(idx % O);
        int n = (int)(idx / O);
        float a = 0.0f;
#pragma unroll
        for (int e = 0; e < EE; ++e) a += emb[n * EE + e] * bp[e * O + o];
        bias[idx] = a;
    }
}

__global__ void zero_f32_kernel(float* __restrict__ p, size_t count)
{
    const size_t stride = (size_t)gridDim.x * blockDim.x;
    for (size_t i = (size_t)blockIdx.x * blockDim.x + threadIdx.x; i < count; i += stride)
        p[i] = 0.0f;
}

// ---------------------------------------------------------------------------
// Kernel 3: xs[n][b][c] = concat(x_t, h) as bf16, node-major.
// xt layout: [B][T][N][Dl].
// ---------------------------------------------------------------------------
__global__ void build_xs_kernel(const float* __restrict__ xt, const float* __restrict__ h,
                                bf16* __restrict__ xs, int Dl, int Cp, int t)
{
    const size_t total  = (size_t)NPAD * BB * Cp;
    const size_t stride = (size_t)gridDim.x * blockDim.x;
    for (size_t idx = (size_t)blockIdx.x * blockDim.x + threadIdx.x; idx < total; idx += stride) {
        int c = (int)(idx % Cp);
        int b = (int)((idx / Cp) % BB);
        int n = (int)(idx / ((size_t)Cp * BB));
        float v = 0.0f;
        if (n < NN) {
            if (c < Dl)           v = xt[(((size_t)b * TT + t) * NN + n) * Dl + c];
            else if (c < Dl + HH) v = h[((size_t)n * BB + b) * HH + (c - Dl)];
        }
        xs[idx] = (bf16)v;
    }
}

// ---------------------------------------------------------------------------
// Kernel 4: aggregation GEMM  Y[n][j] = sum_m A[n][m] * X[m][j]
// A: [MPAD][NPAD] bf16 (async-copied to LDS verbatim), X: [NPAD][J] bf16
// (reg-staged + transposed), Y: [NN][J] bf16.  256 thr = 8 waves (2M x 4J),
// block tile 128x128, K-step 32, double-buffered via ASYNCcnt.
// ---------------------------------------------------------------------------
template<int J>
__global__ __launch_bounds__(256) void agg_gemm_kernel(
    const bf16* __restrict__ A, const bf16* __restrict__ X, bf16* __restrict__ Y)
{
    constexpr int KITERS = NPAD / 32;        // 63
    __shared__ unsigned short As[2][128][40];  // [row][k] + 8-short pad (16B-aligned rows)
    __shared__ unsigned short Bt[2][128][36];  // [j][k] transposed X tile

    const int tid  = threadIdx.x;
    const int wave = tid >> 5;
    const int lane = tid & 31;
    const int mw   = wave >> 2, jw = wave & 3;
    const int bm0  = blockIdx.x * 128;
    const int bj0  = blockIdx.y * 128;
    const int half = lane >> 4, l16 = lane & 15;

    const unsigned int* Xg = (const unsigned int*)X;

    // async A-tile copy: each thread 2 x b128 chunks (per wave: 2 instructions)
    const int ar    = tid >> 1;              // row 0..127
    const int apart = (tid & 1) * 16;        // shorts 0..15 or 16..31
    auto issueA = [&](int k0, int buf) {
#pragma unroll
        for (int c = 0; c < 2; ++c)
            async_b128(lds_off(&As[buf][ar][apart + c * 8]),
                       A + (size_t)(bm0 + ar) * NPAD + k0 + apart + c * 8);
    };
    unsigned xr[8];
    auto loadX = [&](int k0) {
#pragma unroll
        for (int i = 0; i < 8; ++i) {
            int id = tid * 8 + i;
            int r = id >> 6, cp = id & 63;   // k row 0..31, j pair 0..63
            xr[i] = Xg[(size_t)(k0 + r) * (J / 2) + (bj0 >> 1) + cp];
        }
    };
    auto storeX = [&](int buf) {
#pragma unroll
        for (int i = 0; i < 8; ++i) {
            int id = tid * 8 + i;
            int r = id >> 6, cp = id & 63;
            Bt[buf][cp * 2    ][r] = (unsigned short)(xr[i] & 0xffffu);
            Bt[buf][cp * 2 + 1][r] = (unsigned short)(xr[i] >> 16);
        }
    };

    v8f acc[4][2] = {};

    issueA(0, 0);
    loadX(0);
    for (int it = 0; it < KITERS; ++it) {
        const int buf = it & 1;
        __syncthreads();                     // previous compute on this buf done
        storeX(buf);
        if (it + 1 < KITERS) {
            issueA((it + 1) * 32, buf ^ 1);  // overlaps with compute below
            loadX((it + 1) * 32);
            asm volatile("s_wait_asynccnt 2" ::: "memory");  // this buf's asyncs done
        } else {
            asm volatile("s_wait_asynccnt 0" ::: "memory");
        }
        __syncthreads();                     // tiles visible to all waves

        Frag af[4], bfr[2];
#pragma unroll
        for (int fi = 0; fi < 4; ++fi) {
            int row = mw * 64 + fi * 16 + l16;
#pragma unroll
            for (int g = 0; g < 8; ++g) {
                int kb = 2 * (g & 3) + 16 * (g >> 2) + 8 * half;   // ISA A layout
                af[fi].u[g] = *(const unsigned int*)&As[buf][row][kb];
            }
        }
#pragma unroll
        for (int fj = 0; fj < 2; ++fj) {
            int col = jw * 32 + fj * 16 + l16;
#pragma unroll
            for (int g = 0; g < 8; ++g)                            // ISA B layout
                bfr[fj].u[g] = *(const unsigned int*)&Bt[buf][col][2 * (g & 3) + 16 * (g >> 2) + 8 * half];
        }
#pragma unroll
        for (int fi = 0; fi < 4; ++fi)
#pragma unroll
            for (int fj = 0; fj < 2; ++fj)
                acc[fi][fj] = __builtin_amdgcn_wmma_f32_16x16x32_bf16(
                    false, af[fi].v, false, bfr[fj].v, (short)0, acc[fi][fj], false, false);
    }

    // C/D layout: row = vreg + 8*half, col = lane&15.  Full blocks take the
    // unguarded path (const offsets since J is a template param).
    const bool full = (bm0 + 128) <= NN;
#pragma unroll
    for (int fi = 0; fi < 4; ++fi) {
#pragma unroll
        for (int fj = 0; fj < 2; ++fj) {
            const int jcol = bj0 + jw * 32 + fj * 16 + l16;
            const int row0 = bm0 + mw * 64 + fi * 16 + 8 * half;
            bf16* p = Y + (size_t)row0 * J + jcol;
            if (full) {
#pragma unroll
                for (int v = 0; v < 8; ++v) p[(size_t)v * J] = (bf16)acc[fi][fj][v];
            } else {
#pragma unroll
                for (int v = 0; v < 8; ++v)
                    if (row0 + v < NN) p[(size_t)v * J] = (bf16)acc[fi][fj][v];
            }
        }
    }
}

// ---------------------------------------------------------------------------
// Per-node GEMM A-operand (concat(xs[n], xg[n]), 32 x Kq) async copy into LDS.
// ---------------------------------------------------------------------------
template<int Cp, int LDK>
__device__ __forceinline__ void load_aop_async(
    unsigned short (&Aop)[BB][LDK], const bf16* xs, const bf16* xg, int n, int tid)
{
    constexpr int Kq = 2 * Cp;
    constexpr int CH = Kq / 8;               // b128 chunks per row
    for (int id = tid; id < BB * CH; id += 128) {
        int b = id / CH;
        int q = (id % CH) * 8;
        const bf16* src = (q < Cp) ? (xs + ((size_t)n * BB + b) * Cp + q)
                                   : (xg + (size_t)n * (BB * Cp) + b * Cp + (q - Cp));
        async_b128(lds_off(&Aop[b][q]), src);
    }
    asm volatile("s_wait_asynccnt 0" ::: "memory");
}

// ---------------------------------------------------------------------------
// Kernel 5: gate GEMM per node. S = Aop @ W + bias; zr = sigmoid(S);
// z*h -> xs state slot, r -> rbuf.  Grid NN x 128 threads (4 waves).
// ---------------------------------------------------------------------------
template<int Cp, int Dl>
__global__ __launch_bounds__(128) void gate_gemm_kernel(
    bf16* xs,                         // [NPAD][BB][Cp] (read k=0 part, write z*h)
    const bf16* __restrict__ xg,      // [NN][BB*Cp]
    const bf16* __restrict__ W,       // [NN][Kq][128]
    const float* __restrict__ bias,   // [NN][128]
    const float* __restrict__ h,      // [NN][BB][HH]
    float* __restrict__ rbuf)         // [NN][BB][HH]
{
    constexpr int O  = 2 * HH;        // 128
    constexpr int Kq = 2 * Cp;
    const int n = blockIdx.x;
    const int tid = threadIdx.x, wave = tid >> 5, lane = tid & 31;
    const int half = lane >> 4, l16 = lane & 15;
    __shared__ unsigned short Aop[BB][Kq + 8]; // 16B-aligned rows, conflict-free pad
    __shared__ unsigned short Ws[O][36];       // [o][k] tile, transposed

    load_aop_async<Cp>(Aop, xs, xg, n, tid);

    v8f acc[2][2] = {};
    for (int k0 = 0; k0 < Kq; k0 += 32) {
        __syncthreads();
        for (int id = tid; id < 32 * (O >> 1); id += 128) {
            int r  = id / (O >> 1);
            int cp = id % (O >> 1);
            unsigned int v = *(const unsigned int*)&W[((size_t)n * Kq + k0 + r) * O + cp * 2];
            Ws[cp * 2    ][r] = (unsigned short)(v & 0xffffu);
            Ws[cp * 2 + 1][r] = (unsigned short)(v >> 16);
        }
        __syncthreads();

        Frag af[2], bfr[2];
#pragma unroll
        for (int fi = 0; fi < 2; ++fi) {
            int row = fi * 16 + l16;
#pragma unroll
            for (int g = 0; g < 8; ++g) {
                int kb = 2 * (g & 3) + 16 * (g >> 2) + 8 * half;
                af[fi].u[g] = *(const unsigned int*)&Aop[row][k0 + kb];
            }
        }
#pragma unroll
        for (int fj = 0; fj < 2; ++fj) {
            int col = wave * 32 + fj * 16 + l16;
#pragma unroll
            for (int g = 0; g < 8; ++g)
                bfr[fj].u[g] = *(const unsigned int*)&Ws[col][2 * (g & 3) + 16 * (g >> 2) + 8 * half];
        }
#pragma unroll
        for (int fi = 0; fi < 2; ++fi)
#pragma unroll
            for (int fj = 0; fj < 2; ++fj)
                acc[fi][fj] = __builtin_amdgcn_wmma_f32_16x16x32_bf16(
                    false, af[fi].v, false, bfr[fj].v, (short)0, acc[fi][fj], false, false);
    }

#pragma unroll
    for (int fi = 0; fi < 2; ++fi) {
#pragma unroll
        for (int fj = 0; fj < 2; ++fj) {
            int o = wave * 32 + fj * 16 + l16;
            float bv = bias[(size_t)n * O + o];
#pragma unroll
            for (int v = 0; v < 8; ++v) {
                int b = fi * 16 + v + 8 * half;
                float g = sigmoidf_(acc[fi][fj][v] + bv);
                if (o < HH) {   // z: write z*h into the state slot of xs
                    float zh = g * h[((size_t)n * BB + b) * HH + o];
                    xs[((size_t)n * BB + b) * Cp + Dl + o] = (bf16)zh;
                } else {        // r
                    rbuf[((size_t)n * BB + b) * HH + (o - HH)] = g;
                }
            }
        }
    }
}

// ---------------------------------------------------------------------------
// Kernel 6: update GEMM per node, O = H = 64. hc = tanh(S + bias);
// h_new = r*h + (1-r)*hc -> hbuf and output sequence slot.
// ---------------------------------------------------------------------------
template<int Cp>
__global__ __launch_bounds__(128) void upd_gemm_kernel(
    const bf16* __restrict__ xs,      // [NPAD][BB][Cp] (= concat(x_t, z*h))
    const bf16* __restrict__ xg,      // [NN][BB*Cp]
    const bf16* __restrict__ W,       // [NN][Kq][64]
    const float* __restrict__ bias,   // [NN][64]
    const float* __restrict__ rbuf,   // [NN][BB][HH]
    float* h,                          // [NN][BB][HH] (read old, write new)
    float* __restrict__ out,           // [B][T][N][H]
    int t)
{
    constexpr int O  = HH;            // 64
    constexpr int Kq = 2 * Cp;
    const int n = blockIdx.x;
    const int tid = threadIdx.x, wave = tid >> 5, lane = tid & 31;
    const int half = lane >> 4, l16 = lane & 15;
    __shared__ unsigned short Aop[BB][Kq + 8];
    __shared__ unsigned short Ws[O][36];

    load_aop_async<Cp>(Aop, xs, xg, n, tid);

    v8f acc[2] = {};
    for (int k0 = 0; k0 < Kq; k0 += 32) {
        __syncthreads();
        for (int id = tid; id < 32 * (O >> 1); id += 128) {
            int r  = id / (O >> 1);
            int cp = id % (O >> 1);
            unsigned int v = *(const unsigned int*)&W[((size_t)n * Kq + k0 + r) * O + cp * 2];
            Ws[cp * 2    ][r] = (unsigned short)(v & 0xffffu);
            Ws[cp * 2 + 1][r] = (unsigned short)(v >> 16);
        }
        __syncthreads();

        Frag af[2], bfr;
#pragma unroll
        for (int fi = 0; fi < 2; ++fi) {
            int row = fi * 16 + l16;
#pragma unroll
            for (int g = 0; g < 8; ++g) {
                int kb = 2 * (g & 3) + 16 * (g >> 2) + 8 * half;
                af[fi].u[g] = *(const unsigned int*)&Aop[row][k0 + kb];
            }
        }
        {
            int col = wave * 16 + l16;
#pragma unroll
            for (int g = 0; g < 8; ++g)
                bfr.u[g] = *(const unsigned int*)&Ws[col][2 * (g & 3) + 16 * (g >> 2) + 8 * half];
        }
#pragma unroll
        for (int fi = 0; fi < 2; ++fi)
            acc[fi] = __builtin_amdgcn_wmma_f32_16x16x32_bf16(
                false, af[fi].v, false, bfr.v, (short)0, acc[fi], false, false);
    }

    {
        int o = wave * 16 + l16;
        float bv = bias[(size_t)n * O + o];
#pragma unroll
        for (int fi = 0; fi < 2; ++fi) {
#pragma unroll
            for (int v = 0; v < 8; ++v) {
                int b = fi * 16 + v + 8 * half;
                float hc = tanhf(acc[fi][v] + bv);
                size_t hi = ((size_t)n * BB + b) * HH + o;
                float r  = rbuf[hi];
                float hv = h[hi];
                float hn = r * hv + (1.0f - r) * hc;
                h[hi] = hn;
                out[(((size_t)b * TT + t) * NN + n) * HH + o] = hn;
            }
        }
    }
}

// Final state: last[b][n][o] = h[n][b][o]
__global__ void final_h_kernel(const float* __restrict__ h, float* __restrict__ last)
{
    const size_t total  = (size_t)NN * BB * HH;
    const size_t stride = (size_t)gridDim.x * blockDim.x;
    for (size_t idx = (size_t)blockIdx.x * blockDim.x + threadIdx.x; idx < total; idx += stride) {
        int o = (int)(idx % HH);
        int b = (int)((idx / HH) % BB);
        int n = (int)(idx / ((size_t)HH * BB));
        last[((size_t)b * NN + n) * HH + o] = h[idx];
    }
}

// ---------------------------------------------------------------------------
extern "C" void kernel_launch(void* const* d_in, const int* in_sizes, int n_in,
                              void* d_out, int out_size, void* d_ws, size_t ws_size,
                              hipStream_t stream)
{
    (void)in_sizes; (void)n_in; (void)out_size; (void)ws_size;
    const float* x   = (const float*)d_in[0];
    const float* emb = (const float*)d_in[1];
    const float* gw[2] = { (const float*)d_in[2], (const float*)d_in[6] };
    const float* gb[2] = { (const float*)d_in[3], (const float*)d_in[7] };
    const float* uw[2] = { (const float*)d_in[4], (const float*)d_in[8] };
    const float* ub[2] = { (const float*)d_in[5], (const float*)d_in[9] };

    float* out_cur  = (float*)d_out;                       // [B,T,N,H]
    float* out_last = out_cur + (size_t)BB * TT * NN * HH; // [B,N,H]

    char*  ws  = (char*)d_ws;
    size_t off = 0;
    auto take = [&](size_t bytes) -> char* {
        char* p = ws + off;
        off += (bytes + 255) & ~(size_t)255;
        return p;
    };
    bf16*  Abf   = (bf16*) take((size_t)MPAD * NPAD * 2);        // 8.3 MB
    bf16*  xsbuf = (bf16*) take((size_t)NPAD * BB * 128 * 2);    // 16.5 MB (max Cp)
    bf16*  xgbuf = (bf16*) take((size_t)NN * BB * 128 * 2);      // 16.4 MB
    float* hbuf  = (float*)take((size_t)NN * BB * HH * 4);       // 16.4 MB
    float* rbuf  = (float*)take((size_t)NN * BB * HH * 4);       // 16.4 MB
    bf16*  Wg    = (bf16*) take((size_t)NN * 256 * 128 * 2);     // 131 MB (max Kq)
    bf16*  Wu    = (bf16*) take((size_t)NN * 256 * 64 * 2);      // 65.5 MB
    float* biasg = (float*)take((size_t)NN * 128 * 4);
    float* biasu = (float*)take((size_t)NN * 64 * 4);

    adj_softmax_kernel<<<MPAD, 256, 0, stream>>>(emb, Abf);

    for (int l = 0; l < 2; ++l) {
        const int Dl  = (l == 0) ? DIN : HH;
        const int din = Dl + HH;               // 66 or 128
        const int Cp  = (l == 0) ? 80 : 128;   // padded channels
        const float* xt = (l == 0) ? x : out_cur;

        make_w_kernel<<<2048, 256, 0, stream>>>(emb, gw[l], Wg, din, Cp, 2 * HH);
        make_w_kernel<<<2048, 256, 0, stream>>>(emb, uw[l], Wu, din, Cp, HH);
        make_bias_kernel<<<512, 256, 0, stream>>>(emb, gb[l], biasg, 2 * HH);
        make_bias_kernel<<<512, 256, 0, stream>>>(emb, ub[l], biasu, HH);
        zero_f32_kernel<<<2048, 256, 0, stream>>>(hbuf, (size_t)NN * BB * HH);

        for (int t = 0; t < TT; ++t) {
            build_xs_kernel<<<2048, 256, 0, stream>>>(xt, hbuf, xsbuf, Dl, Cp, t);
            if (l == 0) {
                agg_gemm_kernel<BB * 80><<<dim3(MPAD / 128, (BB * 80) / 128), 256, 0, stream>>>(Abf, xsbuf, xgbuf);
                gate_gemm_kernel<80, DIN><<<NN, 128, 0, stream>>>(xsbuf, xgbuf, Wg, biasg, hbuf, rbuf);
                agg_gemm_kernel<BB * 80><<<dim3(MPAD / 128, (BB * 80) / 128), 256, 0, stream>>>(Abf, xsbuf, xgbuf);
                upd_gemm_kernel<80><<<NN, 128, 0, stream>>>(xsbuf, xgbuf, Wu, biasu, rbuf, hbuf, out_cur, t);
            } else {
                agg_gemm_kernel<BB * 128><<<dim3(MPAD / 128, (BB * 128) / 128), 256, 0, stream>>>(Abf, xsbuf, xgbuf);
                gate_gemm_kernel<128, HH><<<NN, 128, 0, stream>>>(xsbuf, xgbuf, Wg, biasg, hbuf, rbuf);
                agg_gemm_kernel<BB * 128><<<dim3(MPAD / 128, (BB * 128) / 128), 256, 0, stream>>>(Abf, xsbuf, xgbuf);
                upd_gemm_kernel<128><<<NN, 128, 0, stream>>>(xsbuf, xgbuf, Wu, biasu, rbuf, hbuf, out_cur, t);
            }
        }
    }
    final_h_kernel<<<2048, 256, 0, stream>>>(hbuf, out_last);
}
